// MultiHeadAttention_V2_29523605193045
// MI455X (gfx1250) — matrix operand order repaired and checked by reading, not compiled, not run on hardware
//
#include <hip/hip_runtime.h>
#include <cstdint>
#include <cstddef>

// ---------------------------------------------------------------------------
// MHA forward for MI455X (gfx1250, wave32, WMMA bf16 -> f32 accumulate)
// B=2, N=2048, D=1024, H=16, HD=64
// ---------------------------------------------------------------------------

#define BATCH 2
#define SEQ   2048
#define DIM   1024
#define HEADS 16
#define HDIM  64
#define BN    (BATCH * SEQ)      // 4096 rows

typedef unsigned int u32x4   __attribute__((ext_vector_type(4)));
typedef float        f32x8   __attribute__((ext_vector_type(8)));
typedef __bf16       bf16x16 __attribute__((ext_vector_type(16)));

union Frag16 {             // 16 bf16 values = one WMMA A/B operand
    bf16x16 b;
    u32x4   q[2];
};

__device__ __forceinline__ unsigned f2bf_u(float f) {
    unsigned u = __float_as_uint(f);
    return (u + 0x7FFFu + ((u >> 16) & 1u)) >> 16;
}
__device__ __forceinline__ uint16_t f2bf(float f) { return (uint16_t)f2bf_u(f); }

__device__ __forceinline__ f32x8 wmma_bf16(const bf16x16& a, const bf16x16& b,
                                           const f32x8& c) {
    return __builtin_amdgcn_wmma_f32_16x16x32_bf16(
        false, a, false, b, (short)0, c, false, false);
}

// ---------------------------------------------------------------------------
// Kernel 1: fp32 -> bf16 elementwise
// ---------------------------------------------------------------------------
__global__ void cvt_bf16_kernel(const float* __restrict__ in,
                                uint16_t* __restrict__ out, int n) {
    int i = blockIdx.x * blockDim.x + threadIdx.x;
    if (i < n) out[i] = f2bf(in[i]);
}

// ---------------------------------------------------------------------------
// Kernel 2: weight fp32 (K x N) -> bf16 transposed (N x K), LDS tiled
// ---------------------------------------------------------------------------
__global__ void wt_transpose_kernel(const float* __restrict__ W,
                                    uint16_t* __restrict__ Wt) {
    __shared__ float tile[32][33];
    int bx = blockIdx.x * 32;   // N-origin of tile
    int by = blockIdx.y * 32;   // K-origin of tile
    int tx = threadIdx.x, ty = threadIdx.y;  // (32, 8)
#pragma unroll
    for (int i = 0; i < 32; i += 8)
        tile[ty + i][tx] = W[(size_t)(by + ty + i) * DIM + bx + tx];
    __syncthreads();
#pragma unroll
    for (int i = 0; i < 32; i += 8)
        Wt[(size_t)(bx + ty + i) * DIM + by + tx] = f2bf(tile[tx][ty + i]);
}

// ---------------------------------------------------------------------------
// Kernel 3: GEMM  out = (A(bf16, 4096x1024) @ Wt^T(stored N x K) + bias)*fscale
//   mode 0: out = per-head row-major bf16  Qh/Kh[(b*H+h)*N + n][hd]
//   mode 1: out = per-head TRANSPOSED bf16 Vt[(b*H+h)*HD + hd][n]
//   mode 2: out = plain fp32 [i][j]
// Block = 128 threads (4 waves); wave computes a 32(M) x 64(N) tile with
// double-buffered fragments so loads overlap the WMMA pipe.
// ---------------------------------------------------------------------------
__global__ __launch_bounds__(128) void gemm_kernel(
    const uint16_t* __restrict__ A, const uint16_t* __restrict__ Wt,
    const float* __restrict__ bias, void* __restrict__ outp, int mode,
    float fscale) {

    const int lane  = threadIdx.x & 31;
    const int wave  = threadIdx.x >> 5;
    const int lh    = lane >> 4;      // half-wave select
    const int ll    = lane & 15;      // lane within half
    const int mbase = blockIdx.y * 128 + wave * 32;
    const int nbase = blockIdx.x * 64;

    f32x8 c[2][4] = {};

    const uint16_t* arow0 = A + (size_t)(mbase + ll) * DIM;
    const uint16_t* arow1 = A + (size_t)(mbase + 16 + ll) * DIM;

    Frag16 af[2][2], bf[2][4];

    auto loadA = [&](Frag16 af_[2], int kk) {
        af_[0].q[0] = *(const u32x4*)(arow0 + kk + lh * 8);
        af_[0].q[1] = *(const u32x4*)(arow0 + kk + 16 + lh * 8);
        af_[1].q[0] = *(const u32x4*)(arow1 + kk + lh * 8);
        af_[1].q[1] = *(const u32x4*)(arow1 + kk + 16 + lh * 8);
    };
    auto loadB = [&](Frag16 bf_[4], int kk) {
#pragma unroll
        for (int t = 0; t < 4; ++t) {
            const uint16_t* p =
                Wt + (size_t)(nbase + 16 * t + ll) * DIM + kk + lh * 8;
            bf_[t].q[0] = *(const u32x4*)(p);
            bf_[t].q[1] = *(const u32x4*)(p + 16);
        }
    };

    loadA(af[0], 0);
    loadB(bf[0], 0);
    int cur = 0;
    for (int kk = 0; kk < DIM; kk += 32) {
        const int nxt = cur ^ 1;
        if (kk + 32 < DIM) {          // issue next step's loads first
            loadA(af[nxt], kk + 32);
            loadB(bf[nxt], kk + 32);
        }
        if (kk + 64 < DIM) {          // CDNA5 global_prefetch_b8 hints
            __builtin_prefetch(arow0 + kk + 64, 0, 0);
            __builtin_prefetch(arow1 + kk + 64, 0, 0);
        }
#pragma unroll
        for (int t = 0; t < 4; ++t) {
            c[0][t] = wmma_bf16(af[cur][0].b, bf[cur][t].b, c[0][t]);
            c[1][t] = wmma_bf16(af[cur][1].b, bf[cur][t].b, c[1][t]);
        }
        cur = nxt;
    }

#pragma unroll
    for (int mt = 0; mt < 2; ++mt) {
#pragma unroll
        for (int t = 0; t < 4; ++t) {
            const int j  = nbase + 16 * t + ll;
            const float bj = bias[j];
#pragma unroll
            for (int r = 0; r < 8; ++r) {
                const int i = mbase + mt * 16 + r + 8 * lh;
                const float v = (c[mt][t][r] + bj) * fscale;
                if (mode == 2) {
                    ((float*)outp)[(size_t)i * DIM + j] = v;
                } else {
                    const int b_ = i >> 11;          // i / SEQ
                    const int n_ = i & (SEQ - 1);
                    const int h_ = j >> 6;           // j / HDIM
                    const int hd = j & (HDIM - 1);
                    const int bh = b_ * HEADS + h_;
                    uint16_t* o = (uint16_t*)outp;
                    if (mode == 0)
                        o[((size_t)bh * SEQ + n_) * HDIM + hd] = f2bf(v);
                    else
                        o[((size_t)bh * HDIM + hd) * SEQ + n_] = f2bf(v);
                }
            }
        }
    }
}

// ---------------------------------------------------------------------------
// Kernel 4: causal flash attention (1/sqrt(D) pre-folded into Q).
//   Qh, Kh : [B*H][N][HD] bf16 row-major
//   Vt     : [B*H][HD][N] bf16 (transposed)
//   Zh out : [B][N][D]    bf16 (merged heads) for the final projection
//
// Scores are computed TRANSPOSED (S^T = K * Q^T): the S^T C-fragment holds
// queries in lanes and keys in VGPRs, which is exactly the A-fragment layout
// required by the P*V WMMA. P is therefore produced by pure in-register
// f32->bf16 packing: no LDS round-trip, no DS waits in the hot loop.
// Softmax row stats live one-query-per-lane; reductions are intra-lane trees
// plus a single cross-half __shfl_xor(.,16).
// ---------------------------------------------------------------------------
__global__ __launch_bounds__(128) void attn_kernel(
    const uint16_t* __restrict__ Qh, const uint16_t* __restrict__ Kh,
    const uint16_t* __restrict__ Vt, uint16_t* __restrict__ Zh) {

    const int lane  = threadIdx.x & 31;
    const int wave  = threadIdx.x >> 5;
    const int lh    = lane >> 4;
    const int ll    = lane & 15;
    const int bh    = blockIdx.y;
    // longest (most-keys) tiles launch first for causal load balance
    const int qtile = (int)gridDim.x - 1 - (int)blockIdx.x;
    const int qbase = qtile * 64 + wave * 16;

    const uint16_t* Qb = Qh + (size_t)bh * SEQ * HDIM;
    const uint16_t* Kb = Kh + (size_t)bh * SEQ * HDIM;
    const uint16_t* Vb = Vt + (size_t)bh * HDIM * SEQ;

    // Q as B-fragments (column = query), two d-steps of 32
    Frag16 bq0, bq1;
    {
        const uint16_t* qrow = Qb + (size_t)(qbase + ll) * HDIM + lh * 8;
        bq0.q[0] = *(const u32x4*)(qrow);
        bq0.q[1] = *(const u32x4*)(qrow + 16);
        bq1.q[0] = *(const u32x4*)(qrow + 32);
        bq1.q[1] = *(const u32x4*)(qrow + 48);
    }

    const int query = qbase + ll;     // this lane's query row
    float mq = -3.0e38f, lq = 0.0f;   // per-query online-softmax stats
    f32x8 z[4] = {};

    Frag16 kf[2][4];   // [buf][tile*2 + dstep]  K as A-fragments
    Frag16 vf[2][4];   // [buf][d-chunk]         V as B-fragments

    auto loadK = [&](Frag16 kf_[4], int kb) {
#pragma unroll
        for (int t = 0; t < 2; ++t) {
            const uint16_t* krow = Kb + (size_t)(kb + 16 * t + ll) * HDIM;
#pragma unroll
            for (int s = 0; s < 2; ++s) {
                kf_[t * 2 + s].q[0] = *(const u32x4*)(krow + 32 * s + lh * 8);
                kf_[t * 2 + s].q[1] =
                    *(const u32x4*)(krow + 32 * s + 16 + lh * 8);
            }
        }
    };
    auto loadV = [&](Frag16 vf_[4], int kb) {
#pragma unroll
        for (int cdx = 0; cdx < 4; ++cdx) {
            const uint16_t* vrow =
                Vb + (size_t)(cdx * 16 + ll) * SEQ + kb + lh * 8;
            vf_[cdx].q[0] = *(const u32x4*)(vrow);
            vf_[cdx].q[1] = *(const u32x4*)(vrow + 16);
        }
    };

    loadK(kf[0], 0);
    loadV(vf[0], 0);
    int cur = 0;

    for (int kb = 0; kb <= qbase; kb += 32) {
        const int nxt = cur ^ 1;
        if (kb + 32 <= qbase) {       // prefetch next key block early
            loadK(kf[nxt], kb + 32);
            loadV(vf[nxt], kb + 32);
        }

        // S^T tiles: rows = keys (VGPRs), cols = queries (lanes)
        f32x8 sT0 = {}, sT1 = {};
        sT0 = wmma_bf16(kf[cur][0].b, bq0.b, sT0);
        sT0 = wmma_bf16(kf[cur][1].b, bq1.b, sT0);
        sT1 = wmma_bf16(kf[cur][2].b, bq0.b, sT1);
        sT1 = wmma_bf16(kf[cur][3].b, bq1.b, sT1);

        // causal mask + per-query (per-lane) online softmax
        float p0[8], p1[8];
        float vmax = -3.0e38f;
#pragma unroll
        for (int r = 0; r < 8; ++r) {
            const int k0 = kb + 8 * lh + r;
            const int k1 = k0 + 16;
            p0[r] = (k0 > query) ? -1.0e30f : sT0[r];
            p1[r] = (k1 > query) ? -1.0e30f : sT1[r];
            vmax = fmaxf(vmax, fmaxf(p0[r], p1[r]));
        }
        vmax = fmaxf(vmax, __shfl_xor(vmax, 16, 32));   // other 16 keys
        const float mnew = fmaxf(mq, vmax);
        const float sc   = __expf(mq - mnew);
        float rs = 0.0f;
#pragma unroll
        for (int r = 0; r < 8; ++r) {
            p0[r] = __expf(p0[r] - mnew);
            p1[r] = __expf(p1[r] - mnew);
            rs += p0[r] + p1[r];
        }
        rs += __shfl_xor(rs, 16, 32);
        lq = lq * sc + rs;
        mq = mnew;

        // rescale Z: broadcast each row's factor from the lane owning it
#pragma unroll
        for (int r = 0; r < 8; ++r) {
            const float scz = __shfl(sc, r + 8 * lh, 32);
            z[0][r] *= scz; z[1][r] *= scz; z[2][r] *= scz; z[3][r] *= scz;
        }

        // pack P (already in A-fragment orientation) to bf16 in registers
        Frag16 pa;
#pragma unroll
        for (int v2 = 0; v2 < 4; ++v2) {
            pa.q[0][v2] = f2bf_u(p0[2 * v2]) | (f2bf_u(p0[2 * v2 + 1]) << 16);
            pa.q[1][v2] = f2bf_u(p1[2 * v2]) | (f2bf_u(p1[2 * v2 + 1]) << 16);
        }

#pragma unroll
        for (int cidx = 0; cidx < 4; ++cidx)
            z[cidx] = wmma_bf16(pa.b, vf[cur][cidx].b, z[cidx]);

        cur = nxt;
    }

    // Normalize and write merged-head bf16 output for the final projection
    const float linv = 1.0f / lq;
    const int b_ = bh >> 4;
    const int h_ = bh & (HEADS - 1);
#pragma unroll
    for (int r = 0; r < 8; ++r) {
        const float lz = __shfl(linv, r + 8 * lh, 32);
        const int q = qbase + r + 8 * lh;
        const size_t base = ((size_t)b_ * SEQ + q) * DIM + h_ * HDIM;
#pragma unroll
        for (int cidx = 0; cidx < 4; ++cidx)
            Zh[base + cidx * 16 + ll] = f2bf(z[cidx][r] * lz);
    }
}

// ---------------------------------------------------------------------------
// Host-side launch
// ---------------------------------------------------------------------------
extern "C" void kernel_launch(void* const* d_in, const int* in_sizes, int n_in,
                              void* d_out, int out_size, void* d_ws,
                              size_t ws_size, hipStream_t stream) {
    const float* X  = (const float*)d_in[0];
    const float* Wq = (const float*)d_in[1];
    const float* bq = (const float*)d_in[2];
    const float* Wk = (const float*)d_in[3];
    const float* bk = (const float*)d_in[4];
    const float* Wv = (const float*)d_in[5];
    const float* bv = (const float*)d_in[6];
    const float* Wo = (const float*)d_in[7];
    const float* bo = (const float*)d_in[8];

    char* w = (char*)d_ws;
    const size_t SZ_X = (size_t)BN * DIM * 2;       // 8 MB
    const size_t SZ_W = (size_t)DIM * DIM * 2;      // 2 MB
    uint16_t* Xb  = (uint16_t*)(w);
    uint16_t* Wqt = (uint16_t*)(w + SZ_X);
    uint16_t* Wkt = (uint16_t*)(w + SZ_X + SZ_W);
    uint16_t* Wvt = (uint16_t*)(w + SZ_X + 2 * SZ_W);
    uint16_t* Wot = (uint16_t*)(w + SZ_X + 3 * SZ_W);
    uint16_t* Qh  = (uint16_t*)(w + SZ_X + 4 * SZ_W);
    uint16_t* Kh  = (uint16_t*)(w + 2 * SZ_X + 4 * SZ_W);
    uint16_t* Vt  = (uint16_t*)(w + 3 * SZ_X + 4 * SZ_W);
    uint16_t* Zh  = (uint16_t*)(w + 4 * SZ_X + 4 * SZ_W);

    // 1) activations -> bf16
    const int nX = BN * DIM;
    cvt_bf16_kernel<<<(nX + 255) / 256, 256, 0, stream>>>(X, Xb, nX);

    // 2) weights -> bf16 transposed (N x K)
    dim3 tb(32, 8), tg(DIM / 32, DIM / 32);
    wt_transpose_kernel<<<tg, tb, 0, stream>>>(Wq, Wqt);
    wt_transpose_kernel<<<tg, tb, 0, stream>>>(Wk, Wkt);
    wt_transpose_kernel<<<tg, tb, 0, stream>>>(Wv, Wvt);
    wt_transpose_kernel<<<tg, tb, 0, stream>>>(Wo, Wot);

    // 3) Q/K/V projections (WMMA GEMMs); 1/sqrt(D)=1/32 folded into Q
    dim3 gg(DIM / 64, BN / 128);
    gemm_kernel<<<gg, 128, 0, stream>>>(Xb, Wqt, bq, Qh, 0, 1.0f / 32.0f);
    gemm_kernel<<<gg, 128, 0, stream>>>(Xb, Wkt, bk, Kh, 0, 1.0f);
    gemm_kernel<<<gg, 128, 0, stream>>>(Xb, Wvt, bv, Vt, 1, 1.0f);

    // 4) causal flash attention
    dim3 ag(SEQ / 64, BATCH * HEADS);
    attn_kernel<<<ag, 128, 0, stream>>>(Qh, Kh, Vt, Zh);

    // 5) output projection (fp32 result)
    gemm_kernel<<<gg, 128, 0, stream>>>(Zh, Wot, bo, d_out, 2, 1.0f);
}
